// RoformerMultiheadAttention_15968688406564
// MI455X (gfx1250) — compile-verified, matrix operand order checked by reading, not run on hardware
//
#include <hip/hip_runtime.h>

#define T_LEN 2048
#define BATCH 2
#define EMB   1024
#define NH    16
#define HD    64
#define BH    (BATCH*NH)   // 32

#ifndef __has_builtin
#define __has_builtin(x) 0
#endif
#if __has_builtin(__builtin_amdgcn_global_load_async_to_lds_b128) && \
    __has_builtin(__builtin_amdgcn_s_wait_asynccnt)
#define HAVE_ASYNC_LDS 1
#else
#define HAVE_ASYNC_LDS 0
#endif

typedef __attribute__((ext_vector_type(16))) __bf16 v16bf;
typedef __attribute__((ext_vector_type(8)))  float  v8f;
typedef __attribute__((ext_vector_type(4)))  int    v4i;
#if HAVE_ASYNC_LDS
typedef __attribute__((address_space(1))) v4i gv4i;   // global 16B vector
typedef __attribute__((address_space(3))) v4i lv4i;   // LDS 16B vector
#endif

union Frag {
    v16bf v;
    __bf16 h[16];
    unsigned short u[16];
    uint4 q[2];
};

// Native fp32 -> bf16 (backend emits v_cvt_*bf16_f32 on gfx1250).
__device__ __forceinline__ __bf16 f2bf(float x) { return (__bf16)x; }

__device__ __forceinline__ unsigned short bf_bits(float x) {
    __bf16 b = (__bf16)x;
    unsigned short s;
    __builtin_memcpy(&s, &b, 2);
    return s;
}

// Intra-wave XOR exchange via ds_swizzle group-of-32 mode: zero VALU setup.
// offset = {xor_mask[4:0] << 10, or_mask=0, and_mask=0x1f}
#define SWZ_XOR(v, m) \
    __int_as_float(__builtin_amdgcn_ds_swizzle(__float_as_int(v), (((m) << 10) | 0x1f)))

__device__ __forceinline__ void pack4(Frag& fr, int base, float4 x) {
    fr.h[base + 0] = f2bf(x.x);
    fr.h[base + 1] = f2bf(x.y);
    fr.h[base + 2] = f2bf(x.z);
    fr.h[base + 3] = f2bf(x.w);
}

// Load 16 bf16 fragment elements for a row: elements 0..7 at row[off0..off0+7],
// elements 8..15 at row[off1..off1+7], converting fp32 -> bf16.
__device__ __forceinline__ v16bf load_f32row_as_bf16(const float* __restrict__ row,
                                                     int off0, int off1) {
    Frag fr;
    pack4(fr, 0,  *(const float4*)(row + off0));
    pack4(fr, 4,  *(const float4*)(row + off0 + 4));
    pack4(fr, 8,  *(const float4*)(row + off1));
    pack4(fr, 12, *(const float4*)(row + off1 + 4));
    return fr.v;
}

// ---------------------------------------------------------------------------
// Kernel 1: QKV projection + scale + RoPE, emits bf16 q/k/v in [bh][t][d].
// One wave per 16x16 tile of the (4096 x 3072) output. K loop over E=1024.
// ---------------------------------------------------------------------------
__global__ __launch_bounds__(256)
void qkv_rope_kernel(const float* __restrict__ X,       // (T,B,E) row-major
                     const float* __restrict__ W,       // (3E, E)
                     const float* __restrict__ bias,    // (3E)
                     const float* __restrict__ relpos,  // (T, HD)
                     unsigned short* __restrict__ qws,
                     unsigned short* __restrict__ kws,
                     unsigned short* __restrict__ vws) {
    const int wave = threadIdx.x >> 5;
    const int lane = threadIdx.x & 31;
    const int half = lane >> 4;
    const int lr   = lane & 15;
    const int NT   = (3 * EMB) / 16;                    // 192 tiles along f
    const int tile = blockIdx.x * 8 + wave;
    const int tn = tile % NT;
    const int tm = tile / NT;

    const float* arow = X + (size_t)(tm * 16 + lr) * EMB;
    const float* brow = W + (size_t)(tn * 16 + lr) * EMB;

    v8f acc = {};
    for (int k0 = 0; k0 < EMB; k0 += 32) {
        const int off0 = k0 + half * 8;
        const int off1 = k0 + 16 + half * 8;
        v16bf a = load_f32row_as_bf16(arow, off0, off1);
        v16bf b = load_f32row_as_bf16(brow, off0, off1);
        acc = __builtin_amdgcn_wmma_f32_16x16x32_bf16(
            false, a, false, b, (short)0, acc, false, false);
    }

    const int f   = tn * 16 + lr;
    const int seg = f >> 10;        // 0=q, 1=k, 2=v  (tile never straddles)
    const int fe  = f & 1023;
    const int h   = fe >> 6;
    const int d   = fe & 63;
    const float bi = bias[f];

#pragma unroll
    for (int r = 0; r < 8; ++r) {
        const int ml = r + half * 8;
        const int m  = tm * 16 + ml;
        const int t  = m >> 1;      // m = t*B + b, B=2
        const int bb = m & 1;
        float val = acc[r] + bi;
        if (seg == 0) val *= 0.125f;            // HD^-0.5
        if (seg <= 1) {                          // RoPE on q and k
            const float sp = relpos[t * HD + (d >> 1)];
            const float cp = relpos[t * HD + 32 + (d >> 1)];
            const float partner = SWZ_XOR(val, 1);          // x[d^1] (lane^1)
            const float rot = (d & 1) ? partner : -partner;
            val = val * cp + rot * sp;
        }
        const unsigned short bits = bf_bits(val);
        const int bhid = bb * NH + h;
        const size_t idx = ((size_t)bhid * T_LEN + t) * HD + d;
        if (seg == 0)      qws[idx] = bits;
        else if (seg == 1) kws[idx] = bits;
        else               vws[idx] = bits;
    }
}

// ---------------------------------------------------------------------------
// Kernel 2: flash attention per head. Block = 8 waves on one bh, 128 q rows.
// K tile staged via async global->LDS (gfx1250 path); V transposed into LDS.
// ---------------------------------------------------------------------------
__global__ __launch_bounds__(256)
void flash_attn_kernel(const unsigned short* __restrict__ qws,
                       const unsigned short* __restrict__ kws,
                       const unsigned short* __restrict__ vws,
                       unsigned short* __restrict__ aws) {  // (T,B,E) bf16
    __shared__ unsigned short Kls[32 * 64];      // [key][d]
    __shared__ unsigned short Vt[64 * 32];       // [d][key]   (transposed)
    __shared__ unsigned short Pst[8][16 * 32];   // per-wave P staging [m][k]

    const int wave = threadIdx.x >> 5;
    const int lane = threadIdx.x & 31;
    const int half = lane >> 4;
    const int lr   = lane & 15;
    const int bh   = blockIdx.x >> 4;
    const int qblk = blockIdx.x & 15;
    const int qrow0 = qblk * 128 + wave * 16;

    const unsigned short* qbase = qws + (size_t)bh * T_LEN * HD;
    const unsigned short* kbase = kws + (size_t)bh * T_LEN * HD;
    const unsigned short* vbase = vws + (size_t)bh * T_LEN * HD;

    // Q fragments: A-layout, two K-chunks (d 0..31, 32..63)
    Frag qa[2];
    {
        const unsigned short* qr = qbase + (size_t)(qrow0 + lr) * HD;
#pragma unroll
        for (int c = 0; c < 2; ++c) {
            qa[c].q[0] = *(const uint4*)(qr + c * 32 + half * 8);
            qa[c].q[1] = *(const uint4*)(qr + c * 32 + 16 + half * 8);
        }
    }

    float m_i[8], l_i[8];
    v8f o[4] = {};
#pragma unroll
    for (int r = 0; r < 8; ++r) { m_i[r] = -1e30f; l_i[r] = 0.0f; }

    for (int kt = 0; kt < T_LEN / 32; ++kt) {
        __syncthreads();
        {   // cooperative staging: 8 bf16 (16B) per thread per tile
            const int i = threadIdx.x;
#if HAVE_ASYNC_LDS
            __builtin_amdgcn_global_load_async_to_lds_b128(
                (gv4i*)(kbase + (size_t)kt * 32 * HD + i * 8),
                (lv4i*)(Kls + i * 8), 0, 0);
#else
            *(uint4*)(Kls + i * 8) =
                *(const uint4*)(kbase + (size_t)kt * 32 * HD + i * 8);
#endif
            const uint4 vv = *(const uint4*)(vbase + (size_t)kt * 32 * HD + i * 8);
            unsigned short tmp[8];
            *(uint4*)tmp = vv;
            const int key = i >> 3;
            const int d0  = (i & 7) * 8;
#pragma unroll
            for (int j = 0; j < 8; ++j) Vt[(d0 + j) * 32 + key] = tmp[j];
#if HAVE_ASYNC_LDS
            __builtin_amdgcn_s_wait_asynccnt(0);
#endif
        }
        __syncthreads();

        // S = Q K^T : two 16-key subtiles, K-dim = HD = 64 (2 WMMAs each)
        v8f s[2];
#pragma unroll
        for (int sub = 0; sub < 2; ++sub) {
            v8f sa = {};
#pragma unroll
            for (int c = 0; c < 2; ++c) {
                Frag kb;
                const unsigned short* kr = Kls + (sub * 16 + lr) * HD + c * 32;
                kb.q[0] = *(const uint4*)(kr + half * 8);
                kb.q[1] = *(const uint4*)(kr + 16 + half * 8);
                sa = __builtin_amdgcn_wmma_f32_16x16x32_bf16(
                    false, qa[c].v, false, kb.v, (short)0, sa, false, false);
            }
            s[sub] = sa;
        }

        // Online softmax update; stage P (bf16) for the second GEMM.
#pragma unroll
        for (int r = 0; r < 8; ++r) {
            float mx = fmaxf(s[0][r], s[1][r]);
            mx = fmaxf(mx, SWZ_XOR(mx, 1));
            mx = fmaxf(mx, SWZ_XOR(mx, 2));
            mx = fmaxf(mx, SWZ_XOR(mx, 4));
            mx = fmaxf(mx, SWZ_XOR(mx, 8));
            const float mn = fmaxf(m_i[r], mx);
            const float alpha = __expf(m_i[r] - mn);
            const float p0 = __expf(s[0][r] - mn);
            const float p1 = __expf(s[1][r] - mn);
            float rs = p0 + p1;
            rs += SWZ_XOR(rs, 1);
            rs += SWZ_XOR(rs, 2);
            rs += SWZ_XOR(rs, 4);
            rs += SWZ_XOR(rs, 8);
            l_i[r] = l_i[r] * alpha + rs;
            m_i[r] = mn;
            const int ml = r + half * 8;
            Pst[wave][ml * 32 + lr]      = bf_bits(p0);
            Pst[wave][ml * 32 + 16 + lr] = bf_bits(p1);
            o[0][r] *= alpha; o[1][r] *= alpha;
            o[2][r] *= alpha; o[3][r] *= alpha;
        }

        // O += P V  (P: 16x32 A-layout from wave-private LDS; V from Vt)
        Frag pa;
        {
            const unsigned short* pr = Pst[wave] + lr * 32;
            pa.q[0] = *(const uint4*)(pr + half * 8);
            pa.q[1] = *(const uint4*)(pr + 16 + half * 8);
        }
#pragma unroll
        for (int c = 0; c < 4; ++c) {
            Frag vb;
            const unsigned short* vr = Vt + (c * 16 + lr) * 32;
            vb.q[0] = *(const uint4*)(vr + half * 8);
            vb.q[1] = *(const uint4*)(vr + 16 + half * 8);
            o[c] = __builtin_amdgcn_wmma_f32_16x16x32_bf16(
                false, pa.v, false, vb.v, (short)0, o[c], false, false);
        }
    }

    // Normalize and scatter into (T,B,E) bf16 context buffer.
    const int b = bh >> 4;
    const int h = bh & 15;
#pragma unroll
    for (int r = 0; r < 8; ++r) {
        const int ml = r + half * 8;
        const int qi = qrow0 + ml;
        const float inv = 1.0f / l_i[r];
        const size_t base = ((size_t)qi * BATCH + b) * EMB + h * HD;
#pragma unroll
        for (int c = 0; c < 4; ++c)
            aws[base + c * 16 + lr] = bf_bits(o[c][r] * inv);
    }
}

// ---------------------------------------------------------------------------
// Kernel 3: output projection. A is bf16 (context), W converted fp32->bf16.
// ---------------------------------------------------------------------------
__global__ __launch_bounds__(256)
void out_proj_kernel(const unsigned short* __restrict__ A,   // (4096,1024) bf16
                     const float* __restrict__ W,            // (1024,1024)
                     const float* __restrict__ bias,
                     float* __restrict__ out) {
    const int wave = threadIdx.x >> 5;
    const int lane = threadIdx.x & 31;
    const int half = lane >> 4;
    const int lr   = lane & 15;
    const int NT   = EMB / 16;                  // 64
    const int tile = blockIdx.x * 8 + wave;
    const int tn = tile % NT;
    const int tm = tile / NT;

    const unsigned short* arow = A + (size_t)(tm * 16 + lr) * EMB;
    const float*          brow = W + (size_t)(tn * 16 + lr) * EMB;

    v8f acc = {};
    for (int k0 = 0; k0 < EMB; k0 += 32) {
        const int off0 = k0 + half * 8;
        const int off1 = k0 + 16 + half * 8;
        Frag a;
        a.q[0] = *(const uint4*)(arow + off0);
        a.q[1] = *(const uint4*)(arow + off1);
        v16bf b = load_f32row_as_bf16(brow, off0, off1);
        acc = __builtin_amdgcn_wmma_f32_16x16x32_bf16(
            false, a.v, false, b, (short)0, acc, false, false);
    }

    const int f = tn * 16 + lr;
    const float bi = bias[f];
#pragma unroll
    for (int r = 0; r < 8; ++r) {
        const int m = tm * 16 + r + half * 8;
        out[(size_t)m * EMB + f] = acc[r] + bi;
    }
}

// ---------------------------------------------------------------------------
extern "C" void kernel_launch(void* const* d_in, const int* in_sizes, int n_in,
                              void* d_out, int out_size, void* d_ws, size_t ws_size,
                              hipStream_t stream) {
    const float* query  = (const float*)d_in[0];   // (T,B,E)
    const float* relpos = (const float*)d_in[1];   // (1,T,HD)
    const float* win    = (const float*)d_in[2];   // (3E,E)
    const float* bin    = (const float*)d_in[3];   // (3E)
    const float* wout   = (const float*)d_in[4];   // (E,E)
    const float* bout   = (const float*)d_in[5];   // (E)
    float* out = (float*)d_out;

    const size_t qkv_elems = (size_t)BH * T_LEN * HD;      // 4M bf16 each
    unsigned short* qws = (unsigned short*)d_ws;
    unsigned short* kws = qws + qkv_elems;
    unsigned short* vws = kws + qkv_elems;
    unsigned short* aws = vws + qkv_elems;                 // context, bf16

    // 4096x3072 GEMM: 256*192 tiles, 8 waves/block
    qkv_rope_kernel<<<(256 * 192) / 8, 256, 0, stream>>>(
        query, win, bin, relpos, qws, kws, vws);

    // 32 heads x 16 q-blocks of 128 rows
    flash_attn_kernel<<<BH * (T_LEN / 128), 256, 0, stream>>>(
        qws, kws, vws, aws);

    // 4096x1024 GEMM: 256*64 tiles, 8 waves/block
    out_proj_kernel<<<(256 * 64) / 8, 256, 0, stream>>>(
        aws, wout, bout, out);
}
